// PressureFlipLinear_15058155339824
// MI455X (gfx1250) — matrix operand order repaired and checked by reading, not compiled, not run on hardware
//
#include <hip/hip_runtime.h>

// ---------------------------------------------------------------------------
// Ternary-weight skinny GEMM for MI455X (gfx1250, wave32, WMMA).
// out[t,o] = sum_k x[t,k] * ternary[o,k] * scales[o*64 + k/128]
// M=32 tokens, N=8192 outputs, K=8192. HBM-bound on the 256MB int32 weights.
// ---------------------------------------------------------------------------

typedef _Float16 v16h __attribute__((ext_vector_type(16)));
typedef _Float16 v8h  __attribute__((ext_vector_type(8)));
typedef float    v8f  __attribute__((ext_vector_type(8)));
typedef int      v4i  __attribute__((ext_vector_type(4)));

constexpr int kOut = 8192;
constexpr int kIn  = 8192;
constexpr int kTok = 32;
constexpr int kGS  = 128;
constexpr int kGroupsPerRow   = kIn / kGS;          // 64
constexpr int kSplit          = 4;                  // K-split for occupancy
constexpr int kChunksPerSplit = kGroupsPerRow / kSplit;  // 16

// Prep: zero the (atomically accumulated) output and convert x -> f16 in ws.
__global__ __launch_bounds__(256) void prep_kernel(const float* __restrict__ x,
                                                   _Float16* __restrict__ xh,
                                                   float* __restrict__ out,
                                                   int n, int do_conv) {
  int i = blockIdx.x * 256 + threadIdx.x;
  if (i < n) {
    out[i] = 0.0f;
    if (do_conv) xh[i] = (_Float16)x[i];
  }
}

template <bool PRECONV>
__global__ __launch_bounds__(32) void ternary_wmma_kernel(
    const void* __restrict__ xin,      // f16 [32][kIn] if PRECONV, else f32
    const int* __restrict__ w,         // [kOut][kIn] ternary int32
    const float* __restrict__ scales,  // [kOut * kGroupsPerRow]
    float* __restrict__ out) {         // [32][kOut]
  const int lane = threadIdx.x;        // 0..31, wave32
  const int n    = lane & 15;          // output column within tile / token row
  const int g1   = lane >> 4;          // lane half selector
  const int o    = blockIdx.x * 16 + n;

  const int kstart = blockIdx.y * (kIn / kSplit);
  const int* wrow  = w + (size_t)o * kIn;

  const _Float16* xh = (const _Float16*)xin;
  const float*    xf = (const float*)xin;

  v8f acc0 = {0.f, 0.f, 0.f, 0.f, 0.f, 0.f, 0.f, 0.f};
  v8f acc1 = {0.f, 0.f, 0.f, 0.f, 0.f, 0.f, 0.f, 0.f};

  for (int gc = 0; gc < kChunksPerSplit; ++gc) {
    const int kg = kstart + gc * kGS;  // chunk base k; one scale group = 128 K
    v8f c0 = {0.f, 0.f, 0.f, 0.f, 0.f, 0.f, 0.f, 0.f};
    v8f c1 = {0.f, 0.f, 0.f, 0.f, 0.f, 0.f, 0.f, 0.f};
#pragma unroll
    for (int s = 0; s < 4; ++s) {
      const int kb = kg + s * 32;

      // B tile (32x16, K x N): half h of lane L holds w[o = L%16][kb + 2*(h&8) + 8*(L>>4) + (h&7)]
      // -> two contiguous 32B int32 runs per lane; streamed nontemporally.
      const int* wp = wrow + kb + 8 * g1;
      v4i w0 = __builtin_nontemporal_load((const v4i*)(wp + 0));
      v4i w1 = __builtin_nontemporal_load((const v4i*)(wp + 4));
      v4i w2 = __builtin_nontemporal_load((const v4i*)(wp + 16));
      v4i w3 = __builtin_nontemporal_load((const v4i*)(wp + 20));
      v16h bm;
#pragma unroll
      for (int j = 0; j < 4; ++j) {
        bm[j]      = (_Float16)w0[j];   // {-1,0,1} exact in f16
        bm[4 + j]  = (_Float16)w1[j];
        bm[8 + j]  = (_Float16)w2[j];
        bm[12 + j] = (_Float16)w3[j];
      }

      // A tiles (16x32, M x K): half h of lane L holds
      // x[t = L%16][kb + 2*(h&8) + 8*(L>>4) + (h&7)] -> two 16B runs per lane.
      v16h am0, am1;
      if constexpr (PRECONV) {
        const _Float16* xp0 = xh + (size_t)n * kIn + kb + 8 * g1;          // tokens 0..15
        const _Float16* xp1 = xh + (size_t)(n + 16) * kIn + kb + 8 * g1;   // tokens 16..31
        v8h a0lo = *(const v8h*)(xp0);
        v8h a0hi = *(const v8h*)(xp0 + 16);
        v8h a1lo = *(const v8h*)(xp1);
        v8h a1hi = *(const v8h*)(xp1 + 16);
        am0 = __builtin_shufflevector(a0lo, a0hi, 0, 1, 2, 3, 4, 5, 6, 7,
                                      8, 9, 10, 11, 12, 13, 14, 15);
        am1 = __builtin_shufflevector(a1lo, a1hi, 0, 1, 2, 3, 4, 5, 6, 7,
                                      8, 9, 10, 11, 12, 13, 14, 15);
      } else {
        const float* xp0 = xf + (size_t)n * kIn + kb + 8 * g1;
        const float* xp1 = xf + (size_t)(n + 16) * kIn + kb + 8 * g1;
#pragma unroll
        for (int j = 0; j < 8; ++j) {
          am0[j]     = (_Float16)xp0[j];
          am0[8 + j] = (_Float16)xp0[16 + j];
          am1[j]     = (_Float16)xp1[j];
          am1[8 + j] = (_Float16)xp1[16 + j];
        }
      }

      // D(16x16 f32): N = lane%16 (= o, fixed per lane), M = v + 8*(lane>>4) (= token)
      c0 = __builtin_amdgcn_wmma_f32_16x16x32_f16(false, am0, false, bm,
                                                  (short)0, c0, false, false);
      c1 = __builtin_amdgcn_wmma_f32_16x16x32_f16(false, am1, false, bm,
                                                  (short)0, c1, false, false);
    }

    // One scale per (o, 128-K group): scalar per lane since o is lane-fixed.
    const int g = kg / kGS;
    const float sgl = scales[o * kGroupsPerRow + g];
#pragma unroll
    for (int v = 0; v < 8; ++v) {
      acc0[v] += c0[v] * sgl;
      acc1[v] += c1[v] * sgl;
    }
  }

  // Accumulate K-split partials into out[t*kOut + o].
#pragma unroll
  for (int v = 0; v < 8; ++v) {
    const int t0 = v + 8 * g1;
    atomicAdd(out + (size_t)t0 * kOut + o, acc0[v]);
    atomicAdd(out + (size_t)(t0 + 16) * kOut + o, acc1[v]);
  }
}

extern "C" void kernel_launch(void* const* d_in, const int* in_sizes, int n_in,
                              void* d_out, int out_size, void* d_ws, size_t ws_size,
                              hipStream_t stream) {
  const float* x      = (const float*)d_in[0];   // [32, 8192] f32
  const int*   tern   = (const int*)d_in[1];     // [8192, 8192] int32 in {-1,0,1}
  const float* scales = (const float*)d_in[2];   // [524288] f32
  float* out = (float*)d_out;                    // [32, 8192] f32

  const int nx = kTok * kIn;  // 262144 (== out element count)
  const bool preconv = ws_size >= (size_t)nx * sizeof(_Float16);
  _Float16* xh = (_Float16*)d_ws;

  prep_kernel<<<dim3((nx + 255) / 256), dim3(256), 0, stream>>>(x, xh, out, nx,
                                                                preconv ? 1 : 0);

  dim3 grid(kOut / 16, kSplit);  // 512 o-tiles x 4 K-splits, 1 wave each
  dim3 block(32);
  if (preconv) {
    ternary_wmma_kernel<true><<<grid, block, 0, stream>>>((const void*)xh, tern,
                                                          scales, out);
  } else {
    ternary_wmma_kernel<false><<<grid, block, 0, stream>>>((const void*)x, tern,
                                                           scales, out);
  }
}
